// LSTMDipNetDecoder_47467978555505
// MI455X (gfx1250) — matrix operand order repaired and checked by reading, not compiled, not run on hardware
//
#include <hip/hip_runtime.h>
#include <math.h>

typedef __attribute__((ext_vector_type(16))) _Float16 v16h;
typedef __attribute__((ext_vector_type(8)))  _Float16 v8h;
typedef __attribute__((ext_vector_type(8)))  float    v8f;

// Problem constants (from reference)
constexpr int Vv = 13042;  // vocab
constexpr int Ll = 81;     // locations
constexpr int Ss = 17;     // steps
constexpr int Bb = 256;    // batch
constexpr int Hh = 200;    // lstm hidden
constexpr int Ee = 80;     // order emb
constexpr int Dd = 240;    // 2*inter emb
constexpr int DE = Dd + Ee;    // 320 (multiple of 32)
constexpr int G4 = 4 * Hh;     // 800
constexpr int Hp = 224;        // Hh padded to multiple of 32
constexpr int Vp = 13056;      // Vv padded to multiple of 16
constexpr int NT_G = G4 / 16;  // 50 N-tiles for gates
constexpr int NT_V2 = Vp / 32; // 408 double-tiles for vocab

// ---------------------------------------------------------------------------
// f16 fragment loads. ISA VGPR layouts (wave32), V_WMMA_F32_16X16X32_F16:
//   A 16x32: M = lane&15; elems 0-7 -> K=ka+0..7, elems 8-15 -> K=16+ka+0..7,
//            ka = (lane>=16)*8
//   B 32x16: N = lane&15; elems 0-7 -> K=kb+0..7, elems 8-15 -> K=16+kb+0..7,
//            kb = (lane>=16)*8
//   D 16x16: vgpr r -> M = r + (lane>=16 ? 8 : 0), N = lane&15
// All K dims padded to multiples of 32 with zeros -> no guards anywhere.
// ---------------------------------------------------------------------------
__device__ __forceinline__ v16h loadA(const _Float16* __restrict__ ap,
                                      int k0, int ka)
{
  const v8h lo = *(const v8h*)(ap + k0 + ka);        // 16B aligned
  const v8h hi = *(const v8h*)(ap + k0 + 16 + ka);   // 16B aligned
  return __builtin_shufflevector(lo, hi, 0, 1, 2, 3, 4, 5, 6, 7,
                                 8, 9, 10, 11, 12, 13, 14, 15);
}
__device__ __forceinline__ v16h loadB(const _Float16* __restrict__ wp,
                                      int k0, int kb)
{
  const v8h lo = *(const v8h*)(wp + k0 + kb);        // 16B aligned
  const v8h hi = *(const v8h*)(wp + k0 + 16 + kb);   // 16B aligned
  return __builtin_shufflevector(lo, hi, 0, 1, 2, 3, 4, 5, 6, 7,
                                 8, 9, 10, 11, 12, 13, 14, 15);
}

// Single-tile: acc += A[m0:m0+16,:K] * W[n0:n0+16,:K]^T (all f16, f32 acc)
template<int K>
__device__ __forceinline__ v8f wmma_tile(v8f acc,
    const _Float16* __restrict__ A, int lda, int m0,
    const _Float16* __restrict__ W, int ldw, int n0)
{
  const int lane = threadIdx.x & 31;
  const int row  = lane & 15;
  const int hi   = (lane & 16) >> 4;
  const int ka   = hi << 3;
  const int kb   = hi << 3;
  const _Float16* ap = A + (size_t)(m0 + row) * lda;
  const _Float16* wp = W + (size_t)(n0 + row) * ldw;
  for (int k0 = 0; k0 < K; k0 += 32) {
    acc = __builtin_amdgcn_wmma_f32_16x16x32_f16(
        false, loadA(ap, k0, ka), false, loadB(wp, k0, kb),
        (short)0, acc, false, false);
  }
  return acc;
}

// Dual-tile: shared A fragment, two adjacent 16-wide N tiles.
template<int K>
__device__ __forceinline__ void wmma_tile2(v8f& acc0, v8f& acc1,
    const _Float16* __restrict__ A, int lda, int m0,
    const _Float16* __restrict__ W, int ldw, int n0)
{
  const int lane = threadIdx.x & 31;
  const int row  = lane & 15;
  const int hi   = (lane & 16) >> 4;
  const int ka   = hi << 3;
  const int kb   = hi << 3;
  const _Float16* ap  = A + (size_t)(m0 + row) * lda;
  const _Float16* wp0 = W + (size_t)(n0 + row) * ldw;
  const _Float16* wp1 = W + (size_t)(n0 + 16 + row) * ldw;
  for (int k0 = 0; k0 < K; k0 += 32) {
    const v16h a = loadA(ap, k0, ka);
    acc0 = __builtin_amdgcn_wmma_f32_16x16x32_f16(
        false, a, false, loadB(wp0, k0, kb), (short)0, acc0, false, false);
    acc1 = __builtin_amdgcn_wmma_f32_16x16x32_f16(
        false, a, false, loadB(wp1, k0, kb), (short)0, acc1, false, false);
  }
}

// ---------------------------------------------------------------------------
// Weight convert + pad: out[r][k] = (r<vrows && k<kin) ? f16(in[r][k]) : 0
// Runs once per launch (weights are step-invariant; 17x amortized).
// ---------------------------------------------------------------------------
__global__ void convert_pad_kernel(const float* __restrict__ in,
                                   _Float16* __restrict__ out,
                                   int vrows, int rows, int kin, int kout)
{
  const int t = blockIdx.x * blockDim.x + threadIdx.x;
  if (t >= rows * kout) return;
  const int r = t / kout, k = t % kout;
  const float v = (r < vrows && k < kin) ? in[(size_t)r * kin + k] : 0.0f;
  out[t] = (_Float16)v;
}

// ---------------------------------------------------------------------------
// Init: c = 0, h16 = 0 (incl. K pad, kept zero forever), oemb = 0, cum_mask = 1
// ---------------------------------------------------------------------------
__global__ void init_kernel(float* c, _Float16* h16, float* oemb,
                            unsigned char* cmask)
{
  const int t = blockIdx.x * blockDim.x + threadIdx.x;
  if (t < Bb * Hh) { c[t] = 0.0f; }
  if (t < Bb * Hp) { h16[t] = (_Float16)0.0f; }
  if (t < Bb * Ee) { oemb[t] = 0.0f; }
  if (t < Bb * Vv) { cmask[t] = 1; }
}

// ---------------------------------------------------------------------------
// Attention + concat:  x16[b] = f16([ a_norm @ enc[b] , order_emb[b] ])
// One block (320 threads) per batch row.
// ---------------------------------------------------------------------------
__global__ void attn_kernel(const float* __restrict__ enc,
                            const int* __restrict__ loc_idxs,
                            const float* __restrict__ MA,
                            const float* __restrict__ oemb,
                            _Float16* __restrict__ x16, int step)
{
  const int b = blockIdx.x;
  const int t = threadIdx.x;
  __shared__ float sel[Ll];
  __shared__ float a[Ll];
  __shared__ float ssum;

  if (t < Ll) {
    const int li = loc_idxs[b * Ll + t];
    sel[t] = (li == step || li == -2) ? 1.0f : 0.0f;
  }
  __syncthreads();
  if (t < Ll) {
    float acc = 0.0f;
    for (int l = 0; l < Ll; ++l) acc += sel[l] * MA[l * Ll + t];
    a[t] = acc;
  }
  __syncthreads();
  if (t == 0) {
    float s = 0.0f;
    for (int j = 0; j < Ll; ++j) s += a[j];
    ssum = s;
  }
  __syncthreads();
  const float inv = (ssum > 0.0f) ? (1.0f / ssum) : 0.0f;

  if (t < Dd) {
    float acc = 0.0f;
    const float* eb = enc + (size_t)b * Ll * Dd + t;
    for (int j = 0; j < Ll; ++j) acc += a[j] * eb[(size_t)j * Dd];
    x16[b * DE + t] = (_Float16)(acc * inv);
  } else if (t < DE) {
    x16[b * DE + t] = (_Float16)oemb[b * Ee + (t - Dd)];
  }
}

// ---------------------------------------------------------------------------
// Gates GEMM: gates = x @ W_ih^T + h @ W_hh^T + b_ih + b_hh   [256, 800]
// One wave per 16x16 tile; 16 x 50 = 800 waves; guard-free f16 operands.
// ---------------------------------------------------------------------------
__global__ void gates_kernel(const _Float16* __restrict__ x16,
                             const _Float16* __restrict__ h16,
                             const _Float16* __restrict__ Wih16,
                             const _Float16* __restrict__ Whh16,
                             const float* __restrict__ b_ih,
                             const float* __restrict__ b_hh,
                             float* __restrict__ gates)
{
  const int wave = blockIdx.x * (blockDim.x >> 5) + (threadIdx.x >> 5);
  const int mt = wave / NT_G, nt = wave % NT_G;
  const int m0 = mt * 16, n0 = nt * 16;
  v8f acc = {};
  acc = wmma_tile<DE>(acc, x16, DE, m0, Wih16, DE, n0);
  acc = wmma_tile<Hp>(acc, h16, Hp, m0, Whh16, Hp, n0);
  const int lane = threadIdx.x & 31;
  const int n = n0 + (lane & 15);
  const float bias = b_ih[n] + b_hh[n];
  const int mbase = m0 + ((lane >> 4) << 3);
  #pragma unroll
  for (int r = 0; r < 8; ++r)
    gates[(mbase + r) * G4 + n] = acc[r] + bias;
}

// ---------------------------------------------------------------------------
// LSTM pointwise: i,f,g,o -> c (fp32), h (f16, padded rows)
// ---------------------------------------------------------------------------
__global__ void lstm_pointwise(const float* __restrict__ gates,
                               _Float16* __restrict__ h16,
                               float* __restrict__ c)
{
  const int t = blockIdx.x * blockDim.x + threadIdx.x;
  if (t >= Bb * Hh) return;
  const int b = t / Hh, j = t % Hh;
  const float* g = gates + b * G4;
  const float gi = 1.0f / (1.0f + __expf(-g[j]));
  const float gf = 1.0f / (1.0f + __expf(-g[Hh + j]));
  const float gg = tanhf(g[2 * Hh + j]);
  const float go = 1.0f / (1.0f + __expf(-g[3 * Hh + j]));
  const float cn = gf * c[t] + gi * gg;
  c[t] = cn;
  h16[b * Hp + j] = (_Float16)(go * tanhf(cn));
}

// ---------------------------------------------------------------------------
// Vocab GEMM: raw scores[b, step, :] = h @ W_out^T + b_out   [256, 13042]
// One wave per 16x32 double-tile (shared A); 16 x 408 = 6528 waves.
// W_out pre-padded to [13056][224] f16 with zero rows -> guard-free loop.
// ---------------------------------------------------------------------------
__global__ void scores_kernel(const _Float16* __restrict__ h16,
                              const _Float16* __restrict__ Wout16,
                              const float* __restrict__ b_out,
                              float* __restrict__ out_scores, int step)
{
  const int wave = blockIdx.x * (blockDim.x >> 5) + (threadIdx.x >> 5);
  const int mt = wave / NT_V2, nt = wave % NT_V2;
  const int m0 = mt * 16, n0 = nt * 32;
  v8f acc0 = {}, acc1 = {};
  wmma_tile2<Hp>(acc0, acc1, h16, Hp, m0, Wout16, Hp, n0);

  const int lane = threadIdx.x & 31;
  const int mbase = m0 + ((lane >> 4) << 3);
  const int na = n0 + (lane & 15);
  const int nb = na + 16;
  if (na < Vv) {
    const float bias = b_out[na];
    #pragma unroll
    for (int r = 0; r < 8; ++r)
      out_scores[(size_t)((mbase + r) * Ss + step) * Vv + na] = acc0[r] + bias;
  }
  if (nb < Vv) {
    const float bias = b_out[nb];
    #pragma unroll
    for (int r = 0; r < 8; ++r)
      out_scores[(size_t)((mbase + r) * Ss + step) * Vv + nb] = acc1[r] + bias;
  }
}

// ---------------------------------------------------------------------------
// Threefry2x32, 20 rounds (jax-style PRNG family) for Gumbel-max sampling
// ---------------------------------------------------------------------------
__device__ __forceinline__ unsigned rotl32(unsigned x, int d) {
  return (x << d) | (x >> (32 - d));
}
__device__ __forceinline__ void threefry2x32(unsigned k0, unsigned k1,
                                             unsigned c0, unsigned c1,
                                             unsigned& o0, unsigned& o1)
{
  const unsigned ks2 = k0 ^ k1 ^ 0x1BD11BDAu;
  unsigned x0 = c0 + k0, x1 = c1 + k1;
  const int RA[4] = {13, 15, 26, 6};
  const int RB[4] = {17, 29, 16, 24};
  #pragma unroll
  for (int i = 0; i < 4; ++i) { x0 += x1; x1 = rotl32(x1, RA[i]); x1 ^= x0; }
  x0 += k1; x1 += ks2 + 1u;
  #pragma unroll
  for (int i = 0; i < 4; ++i) { x0 += x1; x1 = rotl32(x1, RB[i]); x1 ^= x0; }
  x0 += ks2; x1 += k0 + 2u;
  #pragma unroll
  for (int i = 0; i < 4; ++i) { x0 += x1; x1 = rotl32(x1, RA[i]); x1 ^= x0; }
  x0 += k0; x1 += k1 + 3u;
  #pragma unroll
  for (int i = 0; i < 4; ++i) { x0 += x1; x1 = rotl32(x1, RB[i]); x1 ^= x0; }
  x0 += k1; x1 += ks2 + 4u;
  #pragma unroll
  for (int i = 0; i < 4; ++i) { x0 += x1; x1 = rotl32(x1, RA[i]); x1 ^= x0; }
  x0 += ks2; x1 += k0 + 5u;
  o0 = x0; o1 = x1;
}

// ---------------------------------------------------------------------------
// Mask scores in place, Gumbel-max categorical per batch row, update
// cum_mask + order_emb + out idxs. One 256-thread block per batch row.
// ---------------------------------------------------------------------------
__global__ void sample_kernel(float* __restrict__ out_scores,
                              const unsigned char* __restrict__ order_masks,
                              unsigned char* __restrict__ cmask,
                              const float* __restrict__ order_embedding,
                              float* __restrict__ oemb,
                              int* __restrict__ out_idx, int step)
{
  const int b = blockIdx.x;
  const int t = threadIdx.x;
  float* sc = out_scores + (size_t)(b * Ss + step) * Vv;
  const unsigned char* om = order_masks + (size_t)(b * Ss + step) * Vv;
  unsigned char* cm = cmask + (size_t)b * Vv;

  const unsigned k0 = 1234u;
  const unsigned k1 = 0x9E3779B9u * (unsigned)step + 0xBB67AE85u;

  float bestv = -INFINITY;
  int besti = 0;
  for (int v = t; v < Vv; v += 256) {
    const bool eff = (om[v] != 0) && (cm[v] != 0);
    const float masked = fminf(sc[v], eff ? 9.0e8f : -1.0e8f);
    sc[v] = masked;  // reference returns post-mask scores
    unsigned r0, r1;
    threefry2x32(k0, k1, (unsigned)v, ((unsigned)b << 8) ^ 0xA5u, r0, r1);
    const float u = (float)(r0 >> 8) * (1.0f / 16777216.0f);
    const float gum = -__logf(-__logf(u + 1e-12f) + 1e-12f);
    const float val = masked + gum;  // TEMPERATURE == 1
    if (val > bestv) { bestv = val; besti = v; }
  }

  __shared__ float sv[256];
  __shared__ int si[256];
  sv[t] = bestv; si[t] = besti;
  __syncthreads();
  for (int s2 = 128; s2 > 0; s2 >>= 1) {
    if (t < s2 && sv[t + s2] > sv[t]) { sv[t] = sv[t + s2]; si[t] = si[t + s2]; }
    __syncthreads();
  }
  const int chosen = si[0];
  if (t == 0) {
    out_idx[b * Ss + step] = chosen;
    cm[chosen] = 0;  // exclude for future steps (~eye compat table)
  }
  if (t < Ee) {
    oemb[b * Ee + t] = order_embedding[(size_t)chosen * Ee + t];
  }
}

// ---------------------------------------------------------------------------
// Host launcher: weight-convert + init, then 17 sequential decode steps.
// ---------------------------------------------------------------------------
extern "C" void kernel_launch(void* const* d_in, const int* in_sizes, int n_in,
                              void* d_out, int out_size, void* d_ws, size_t ws_size,
                              hipStream_t stream)
{
  const float* enc             = (const float*)d_in[0];
  // d_in[1] in_adj_phase: unused by reference forward
  const int* loc_idxs          = (const int*)d_in[2];
  const unsigned char* omasks  = (const unsigned char*)d_in[3];
  const float* order_embedding = (const float*)d_in[4];
  const float* W_ih            = (const float*)d_in[5];
  const float* W_hh            = (const float*)d_in[6];
  const float* b_ih            = (const float*)d_in[7];
  const float* b_hh            = (const float*)d_in[8];
  const float* W_out           = (const float*)d_in[9];
  const float* b_out           = (const float*)d_in[10];
  const float* MA              = (const float*)d_in[11];

  // Workspace layout; every sub-buffer size is a multiple of 32B. ~11.4 MB.
  char* ws = (char*)d_ws;
  _Float16* x16    = (_Float16*)ws;                 ws += Bb * DE * 2;   // [256,320]
  _Float16* h16    = (_Float16*)ws;                 ws += Bb * Hp * 2;   // [256,224]
  float*    cbuf   = (float*)ws;                    ws += Bb * Hh * 4;   // [256,200]
  float*    gbuf   = (float*)ws;                    ws += Bb * G4 * 4;   // [256,800]
  float*    oemb   = (float*)ws;                    ws += Bb * Ee * 4;   // [256,80]
  _Float16* Wih16  = (_Float16*)ws;                 ws += G4 * DE * 2;   // [800,320]
  _Float16* Whh16  = (_Float16*)ws;                 ws += G4 * Hp * 2;   // [800,224]
  _Float16* Wout16 = (_Float16*)ws;                 ws += (size_t)Vp * Hp * 2; // [13056,224]
  unsigned char* cmask = (unsigned char*)ws;                             // [256,13042]

  int* out_idx      = (int*)d_out;             // [256,17] int32 region
  float* out_scores = (float*)d_out + Bb * Ss; // [256,17,13042] fp32

  // One-time per launch: state init + weight conversion (deterministic).
  init_kernel<<<(Bb * Vv + 255) / 256, 256, 0, stream>>>(cbuf, h16, oemb, cmask);
  convert_pad_kernel<<<(G4 * DE + 255) / 256, 256, 0, stream>>>(
      W_ih, Wih16, G4, G4, DE, DE);
  convert_pad_kernel<<<(G4 * Hp + 255) / 256, 256, 0, stream>>>(
      W_hh, Whh16, G4, G4, Hh, Hp);
  convert_pad_kernel<<<(Vp * Hp + 255) / 256, 256, 0, stream>>>(
      W_out, Wout16, Vv, Vp, Hh, Hp);

  for (int step = 0; step < Ss; ++step) {
    attn_kernel<<<Bb, DE, 0, stream>>>(enc, loc_idxs, MA, oemb, x16, step);
    gates_kernel<<<(16 * NT_G) / 8, 256, 0, stream>>>(x16, h16, Wih16, Whh16,
                                                      b_ih, b_hh, gbuf);
    lstm_pointwise<<<(Bb * Hh + 255) / 256, 256, 0, stream>>>(gbuf, h16, cbuf);
    scores_kernel<<<(16 * NT_V2) / 8, 256, 0, stream>>>(h16, Wout16, b_out,
                                                        out_scores, step);
    sample_kernel<<<Bb, 256, 0, stream>>>(out_scores, omasks, cmask,
                                          order_embedding, oemb, out_idx, step);
  }
}
